// MultiHeadAttention_30459908063647
// MI455X (gfx1250) — matrix operand order repaired and checked by reading, not compiled, not run on hardware
//
#include <hip/hip_runtime.h>

#define DEV __device__ __forceinline__

typedef __attribute__((ext_vector_type(16))) __bf16 v16bf;
typedef __attribute__((ext_vector_type(8)))  __bf16 v8bf;
typedef __attribute__((ext_vector_type(8)))  float  v8f;

typedef __attribute__((ext_vector_type(4))) unsigned int u32x4;
typedef __attribute__((ext_vector_type(8))) int          i32x8;
typedef __attribute__((ext_vector_type(4))) int          i32x4;

static constexpr int BATCH  = 2;
static constexpr int SEQ    = 2048;
static constexpr int DMODEL = 1024;
static constexpr int NHEAD  = 16;
static constexpr int HDIM   = 64;
static constexpr int MTOT   = BATCH * SEQ;   // 4096 rows total

// Padded LDS row strides (TDM pad feature): K rows 64 -> 72 elems, V 32 -> 40.
static constexpr int KT_LDS_STRIDE = 72;
static constexpr int VT_LDS_STRIDE = 40;

#if defined(__AMDGCN__) && __has_builtin(__builtin_amdgcn_tensor_load_to_lds) && \
    __has_builtin(__builtin_amdgcn_s_wait_tensorcnt)
#define USE_TDM 1
#else
#define USE_TDM 0
#endif

#if __has_include(<hip/amd_detail/amd_gfx1250_TDM.h>)
#define TDM_SIX_ARGS 1
#else
#define TDM_SIX_ARGS 0
#endif

// ---------------------------------------------------------------- WMMA helpers

DEV v8f wmma_bf16(v16bf a, v16bf b, v8f c) {
  // v_wmma_f32_16x16x32_bf16  (neg_a, A, neg_b, B, c_mod, C, reuse_a, reuse_b)
  return __builtin_amdgcn_wmma_f32_16x16x32_bf16(false, a, false, b, (short)0, c,
                                                 false, false);
}

DEV v8f vzero8() {
  v8f z;
#pragma unroll
  for (int i = 0; i < 8; ++i) z[i] = 0.0f;
  return z;
}

// 16 contiguous bf16 (32 bytes) as two 16B halves (works for 16B alignment).
DEV v16bf ld32B(const __bf16* p) {
  union { v16bf v; v8bf h[2]; } u;
  u.h[0] = *(const v8bf*)(p);
  u.h[1] = *(const v8bf*)(p + 8);
  return u.v;
}

// A fragment (16x32, row-major, leading dim ld), base points at tile origin.
// Contiguous k order used here (B matches): lane L holds row m=L%16; k = [half*16, half*16+16)
// where half = L/16.
DEV v16bf load_a_frag(const __bf16* a, int ld, int lane) {
  const int m  = lane & 15;
  const int hl = lane >> 4;
  const __bf16* p = a + (size_t)m * ld + hl * 16;
  union { v16bf v; v8bf h[2]; } u;
  u.h[0] = *(const v8bf*)(p);
  u.h[1] = *(const v8bf*)(p + 8);
  return u.v;
}

// B fragment (32x16) from K-contiguous storage: element(k,n) = bt[n*ld + k].
// Same contiguous k order as A: lanes 0-15 hold K=0..15, lanes 16-31 hold K=16..31.
DEV v16bf load_b_frag_t(const __bf16* bt, int ld, int lane) {
  const int n  = lane & 15;
  const int kb = (lane >> 4) << 4;
  return *(const v16bf*)(bt + (size_t)n * ld + kb);
}

// ---------------------------------------------------------------- TDM helper
#if USE_TDM
// 2D tile DMA global -> LDS.  tile_w/tile_h in elements (2B each),
// stride_elems = tensor dim0 stride.  LDS rows padded via TDM pad feature.
DEV void tdm_load_2d(const void* gptr, unsigned lds_off, unsigned tile_w,
                     unsigned tile_h, unsigned stride_elems,
                     unsigned pad_interval_code, unsigned pad_amount_code) {
  unsigned long long ga = (unsigned long long)(size_t)gptr;
  u32x4 g0;
  g0[0] = 1u;                                           // count=1, user mode
  g0[1] = lds_off;                                      // lds_addr (bytes)
  g0[2] = (unsigned)(ga & 0xffffffffull);               // global_addr[31:0]
  g0[3] = (unsigned)((ga >> 32) & 0x01ffffffull)        // global_addr[56:32]
          | (2u << 30);                                 // type = 2 ("image")
  i32x8 g1;
  unsigned dw0 = (1u << 16)                             // data_size = 2 bytes
               | (1u << 20)                             // pad_enable
               | (pad_interval_code << 22)              // pad_interval
               | (pad_amount_code << 25);               // pad_amount (dwords-1)
  g1[0] = (int)dw0;
  g1[1] = (int)((tile_w & 0xffffu) << 16);              // tensor_dim0 lo16
  g1[2] = (int)(((tile_w >> 16) & 0xffffu) |
                ((tile_h & 0xffffu) << 16));            // dim0 hi16 | dim1 lo16
  g1[3] = (int)(((tile_h >> 16) & 0xffffu) |
                ((tile_w & 0xffffu) << 16));            // dim1 hi16 | tile_dim0
  g1[4] = (int)(tile_h & 0xffffu);                      // tile_dim1, tile_dim2=0
  g1[5] = (int)stride_elems;                            // dim0_stride lo32
  g1[6] = 0;                                            // stride hi16, dim1_stride=0
  g1[7] = 0;
  i32x4 gz;
  gz[0] = gz[1] = gz[2] = gz[3] = 0;                    // groups 2/3: 2D tensor
#if TDM_SIX_ARGS
  i32x8 gz8;
#pragma unroll
  for (int i = 0; i < 8; ++i) gz8[i] = 0;
  __builtin_amdgcn_tensor_load_to_lds(g0, g1, gz, gz, gz8, 0);
#else
  __builtin_amdgcn_tensor_load_to_lds(g0, g1, gz, gz, 0);
#endif
}
#endif  // USE_TDM

// ---------------------------------------------------------------- prep kernels

__global__ void f32_to_bf16(const float* __restrict__ in,
                            __bf16* __restrict__ out, size_t n) {
  size_t i = (size_t)blockIdx.x * 256 + threadIdx.x;
  if (i < n) out[i] = (__bf16)in[i];
}

// W [K=1024][N=1024] f32 -> Wt[N][K] bf16 (K-contiguous for B fragments)
__global__ void conv_transpose_w(const float* __restrict__ W,
                                 __bf16* __restrict__ Wt) {
  size_t i = (size_t)blockIdx.x * 256 + threadIdx.x;
  if (i >= (size_t)DMODEL * DMODEL) return;
  int k = (int)(i % DMODEL);
  int n = (int)(i / DMODEL);
  Wt[i] = (__bf16)W[(size_t)k * DMODEL + n];
}

// V [b][n][h*64+d] bf16 -> Vt [b][h][d][n] bf16 (n-contiguous for PV B frags)
__global__ void transpose_v(const __bf16* __restrict__ V,
                            __bf16* __restrict__ Vt) {
  size_t idx = (size_t)blockIdx.x * 256 + threadIdx.x;
  if (idx >= (size_t)MTOT * DMODEL) return;
  int c = (int)(idx % DMODEL);
  size_t bn = idx / DMODEL;
  int n = (int)(bn % SEQ);
  int b = (int)(bn / SEQ);
  int h = c / HDIM, d = c % HDIM;
  Vt[(((size_t)b * NHEAD + h) * HDIM + d) * SEQ + n] = V[idx];
}

// ---------------------------------------------------------------- GEMM (WMMA)
// One wave computes a 32(M) x 64(N) tile of C = A[M,K] @ Bt^T (+bias).
template <bool FINAL>
__global__ __launch_bounds__(32) void gemm_bf16_wmma(
    const __bf16* __restrict__ A, const __bf16* __restrict__ Bt,
    const float* __restrict__ bias, __bf16* __restrict__ Ob,
    float* __restrict__ Of, int M, int N, int K) {
  const int lane = threadIdx.x;
  const int m0 = blockIdx.x * 32;
  const int n0 = blockIdx.y * 64;

  v8f acc[2][4];
#pragma unroll
  for (int i = 0; i < 2; ++i)
#pragma unroll
    for (int t = 0; t < 4; ++t) acc[i][t] = vzero8();

  for (int k0 = 0; k0 < K; k0 += 32) {
#if defined(__AMDGCN__)
    if (k0 + 64 < K) {  // prefetch next-next K-step lines (global_prefetch_b8)
      __builtin_prefetch(A + (size_t)(m0 + (lane & 15)) * K + k0 + 64, 0, 0);
      __builtin_prefetch(Bt + (size_t)(n0 + lane) * K + k0 + 64, 0, 0);
    }
#endif
    v16bf a0 = load_a_frag(A + (size_t)m0 * K + k0, K, lane);
    v16bf a1 = load_a_frag(A + (size_t)(m0 + 16) * K + k0, K, lane);
#pragma unroll
    for (int t = 0; t < 4; ++t) {
      v16bf bf = load_b_frag_t(Bt + (size_t)(n0 + t * 16) * K + k0, K, lane);
      acc[0][t] = wmma_bf16(a0, bf, acc[0][t]);
      acc[1][t] = wmma_bf16(a1, bf, acc[1][t]);
    }
  }

  const int hl = lane >> 4;
  const int nl = lane & 15;
#pragma unroll
  for (int i = 0; i < 2; ++i)
#pragma unroll
    for (int t = 0; t < 4; ++t)
#pragma unroll
      for (int r = 0; r < 8; ++r) {
        int m = m0 + i * 16 + r + 8 * hl;       // C/D layout: m = r + 8*(L/16)
        int nn = n0 + t * 16 + nl;              // n = L%16
        float v = acc[i][t][r];
        if (FINAL)
          Of[(size_t)m * N + nn] = v + bias[nn];
        else
          Ob[(size_t)m * N + nn] = (__bf16)v;
      }
}

// ---------------------------------------------------------------- attention
// One wave handles one (batch, head, 16-query tile); flash-style online
// softmax over key blocks of 32; causal masking.  K/V tiles are staged into
// LDS by the Tensor Data Mover (double-buffered) when available.
__global__ __launch_bounds__(32) void attn_wmma(
    const __bf16* __restrict__ Q, const __bf16* __restrict__ Km,
    const __bf16* __restrict__ Vt, __bf16* __restrict__ ctx) {
  const int lane = threadIdx.x;
  const int qt = blockIdx.x;               // 0..SEQ/16-1
  const int bh = blockIdx.y;               // 0..BATCH*NHEAD-1
  const int b = bh >> 4, h = bh & 15;
  const int nl = lane & 15;
  const int hl = lane >> 4;
  const float SCALE = 0.125f;              // 1/sqrt(64)
  const float L2E = 1.4426950408889634f;
  const float NEG = -3.0e38f;

  const __bf16* Qbase = Q + ((size_t)(b * SEQ + qt * 16)) * DMODEL + h * HDIM;
  const __bf16* Kbase = Km + (size_t)b * SEQ * DMODEL + h * HDIM;
  const __bf16* Vbase = Vt + ((size_t)(b * NHEAD + h)) * HDIM * SEQ;

  v16bf qa[2];
  qa[0] = load_a_frag(Qbase, DMODEL, lane);        // k = 0..31 of head dim
  qa[1] = load_a_frag(Qbase + 32, DMODEL, lane);   // k = 32..63

  v8f o[4];
  float rmax[8], rsum[8];
#pragma unroll
  for (int t = 0; t < 4; ++t) o[t] = vzero8();
#pragma unroll
  for (int r = 0; r < 8; ++r) { rmax[r] = NEG; rsum[r] = 0.0f; }

  __shared__ __attribute__((aligned(16))) __bf16 Pl[16 * 32];

  const int nb = qt / 2 + 1;                // causal: keys up to qt*16+15

#if USE_TDM
  // Double-buffered K (32 keys x 64 d) and V (64 d x 32 n) tiles, rows padded
  // by TDM to kill LDS bank conflicts on the strided fragment reads.
  __shared__ __attribute__((aligned(32))) __bf16 ktile[2][32 * KT_LDS_STRIDE];
  __shared__ __attribute__((aligned(32))) __bf16 vtile[2][64 * VT_LDS_STRIDE];
  // Preload block 0.  K rows: 64 elems (32 dw) -> pad_interval code 4 (32 dw),
  // pad_amount code 3 (4 dw = 8 elems).  V rows: 32 elems (16 dw) -> code 3/3.
  tdm_load_2d(Kbase, (unsigned)(size_t)&ktile[0][0], 64, 32, DMODEL, 4, 3);
  tdm_load_2d(Vbase, (unsigned)(size_t)&vtile[0][0], 32, 64, SEQ, 3, 3);
#endif

  for (int kb = 0; kb < nb; ++kb) {
    const int kbase = kb * 32;
    const int buf = kb & 1;

#if USE_TDM
    if (kb + 1 < nb) {
      const int nbuf = buf ^ 1;
      const int nkbase = kbase + 32;
      tdm_load_2d(Kbase + (size_t)nkbase * DMODEL,
                  (unsigned)(size_t)&ktile[nbuf][0], 64, 32, DMODEL, 4, 3);
      tdm_load_2d(Vbase + nkbase,
                  (unsigned)(size_t)&vtile[nbuf][0], 32, 64, SEQ, 3, 3);
      __builtin_amdgcn_s_wait_tensorcnt(2);   // in-order: current pair done
    } else {
      __builtin_amdgcn_s_wait_tensorcnt(0);
    }
#else
    (void)buf;
#endif

    // ---- scores S[16x32] = Q[16x64] @ K^T : 4 WMMAs
    v8f s[2];
    s[0] = vzero8();
    s[1] = vzero8();
#pragma unroll
    for (int st = 0; st < 2; ++st) {
#if USE_TDM
      const __bf16* krow =
          &ktile[buf][(size_t)(st * 16 + nl) * KT_LDS_STRIDE + (hl << 4)];
      v16bf kf0 = ld32B(krow);         // k 0..31 of head dim
      v16bf kf1 = ld32B(krow + 32);    // k 32..63
#else
      const __bf16* kp =
          Kbase + (size_t)(kbase + st * 16 + nl) * DMODEL + (hl << 4);
      v16bf kf0 = *(const v16bf*)(kp);
      v16bf kf1 = *(const v16bf*)(kp + 32);
#endif
      s[st] = wmma_bf16(qa[0], kf0, s[st]);
      s[st] = wmma_bf16(qa[1], kf1, s[st]);
    }

    // ---- scale, causal mask, online softmax (rows split across lane halves)
#pragma unroll
    for (int r = 0; r < 8; ++r) {
      const int q_idx = qt * 16 + r + 8 * hl;
      float x0 = s[0][r] * SCALE;
      float x1 = s[1][r] * SCALE;
      if (kbase + nl > q_idx) x0 = NEG;
      if (kbase + 16 + nl > q_idx) x1 = NEG;
      float mx = fmaxf(x0, x1);
      mx = fmaxf(mx, __shfl_xor(mx, 1, 32));
      mx = fmaxf(mx, __shfl_xor(mx, 2, 32));
      mx = fmaxf(mx, __shfl_xor(mx, 4, 32));
      mx = fmaxf(mx, __shfl_xor(mx, 8, 32));
      float mnew = fmaxf(rmax[r], mx);
      float corr = exp2f((rmax[r] - mnew) * L2E);
      float p0 = exp2f((x0 - mnew) * L2E);
      float p1 = exp2f((x1 - mnew) * L2E);
      float sp = p0 + p1;
      sp += __shfl_xor(sp, 1, 32);
      sp += __shfl_xor(sp, 2, 32);
      sp += __shfl_xor(sp, 4, 32);
      sp += __shfl_xor(sp, 8, 32);
      rsum[r] = rsum[r] * corr + sp;
      rmax[r] = mnew;
#pragma unroll
      for (int t = 0; t < 4; ++t) o[t][r] *= corr;
      s[0][r] = p0;   // keep probabilities for LDS transpose
      s[1][r] = p1;
    }

    // ---- transpose P from C/D layout to A layout via LDS (row-major 16x32)
    __syncthreads();  // WAR guard vs previous iteration's reads (1 wave)
#pragma unroll
    for (int r = 0; r < 8; ++r) {
      int m = r + 8 * hl;
      Pl[m * 32 + nl] = (__bf16)s[0][r];
      Pl[m * 32 + 16 + nl] = (__bf16)s[1][r];
    }
    __syncthreads();
    union { v16bf v; v8bf hh[2]; } pa;
    pa.hh[0] = *(const v8bf*)&Pl[nl * 32 + hl * 16];
    pa.hh[1] = *(const v8bf*)&Pl[nl * 32 + hl * 16 + 8];

    // ---- O += P[16x32] @ V[32x64] : 4 WMMAs (V is d-major: n-contiguous)
#pragma unroll
    for (int t = 0; t < 4; ++t) {
#if USE_TDM
      const __bf16* vrow =
          &vtile[buf][(size_t)(t * 16 + nl) * VT_LDS_STRIDE + (hl << 4)];
      v16bf vf = ld32B(vrow);
#else
      const __bf16* vp =
          Vbase + (size_t)(t * 16 + nl) * SEQ + kbase + (hl << 4);
      v16bf vf = *(const v16bf*)vp;
#endif
      o[t] = wmma_bf16(pa.v, vf, o[t]);
    }
  }

  // ---- normalize and store ctx (bf16, [b][n][h*64+d])
#pragma unroll
  for (int r = 0; r < 8; ++r) {
    float inv = 1.0f / rsum[r];
    int m = qt * 16 + r + 8 * hl;
#pragma unroll
    for (int t = 0; t < 4; ++t) {
      ctx[((size_t)(b * SEQ) + m) * DMODEL + h * HDIM + t * 16 + nl] =
          (__bf16)(o[t][r] * inv);
    }
  }
}

// ---------------------------------------------------------------- launch

extern "C" void kernel_launch(void* const* d_in, const int* in_sizes, int n_in,
                              void* d_out, int out_size, void* d_ws,
                              size_t ws_size, hipStream_t stream) {
  (void)in_sizes; (void)n_in; (void)out_size; (void)ws_size;

  const float* x  = (const float*)d_in[0];
  const float* Wq = (const float*)d_in[1];
  const float* Wk = (const float*)d_in[2];
  const float* Wv = (const float*)d_in[3];
  const float* Wo = (const float*)d_in[4];
  const float* bo = (const float*)d_in[5];
  float* out = (float*)d_out;

  char* ws = (char*)d_ws;
  auto alloc = [&](size_t bytes) -> void* {
    void* p = (void*)ws;
    ws += (bytes + 255) & ~(size_t)255;
    return p;
  };
  const size_t XN = (size_t)MTOT * DMODEL;     // 4096*1024
  const size_t WN = (size_t)DMODEL * DMODEL;   // 1024*1024

  __bf16* xb   = (__bf16*)alloc(XN * 2);
  __bf16* Wqt  = (__bf16*)alloc(WN * 2);
  __bf16* Wkt  = (__bf16*)alloc(WN * 2);
  __bf16* Wvt  = (__bf16*)alloc(WN * 2);
  __bf16* Wot  = (__bf16*)alloc(WN * 2);
  __bf16* Qb   = (__bf16*)alloc(XN * 2);
  __bf16* Kb   = (__bf16*)alloc(XN * 2);
  __bf16* Vb   = (__bf16*)alloc(XN * 2);
  __bf16* Vtb  = (__bf16*)alloc(XN * 2);
  __bf16* ctxb = (__bf16*)alloc(XN * 2);

  const int T = 256;
  f32_to_bf16<<<(unsigned)((XN + T - 1) / T), T, 0, stream>>>(x, xb, XN);
  conv_transpose_w<<<(unsigned)((WN + T - 1) / T), T, 0, stream>>>(Wq, Wqt);
  conv_transpose_w<<<(unsigned)((WN + T - 1) / T), T, 0, stream>>>(Wk, Wkt);
  conv_transpose_w<<<(unsigned)((WN + T - 1) / T), T, 0, stream>>>(Wv, Wvt);
  conv_transpose_w<<<(unsigned)((WN + T - 1) / T), T, 0, stream>>>(Wo, Wot);

  dim3 gg(MTOT / 32, DMODEL / 64);
  gemm_bf16_wmma<false><<<gg, 32, 0, stream>>>(xb, Wqt, nullptr, Qb, nullptr,
                                               MTOT, DMODEL, DMODEL);
  gemm_bf16_wmma<false><<<gg, 32, 0, stream>>>(xb, Wkt, nullptr, Kb, nullptr,
                                               MTOT, DMODEL, DMODEL);
  gemm_bf16_wmma<false><<<gg, 32, 0, stream>>>(xb, Wvt, nullptr, Vb, nullptr,
                                               MTOT, DMODEL, DMODEL);

  transpose_v<<<(unsigned)((XN + T - 1) / T), T, 0, stream>>>(Vb, Vtb);

  dim3 ga(SEQ / 16, BATCH * NHEAD);
  attn_wmma<<<ga, 32, 0, stream>>>(Qb, Kb, Vtb, ctxb);

  gemm_bf16_wmma<true><<<gg, 32, 0, stream>>>(ctxb, Wot, bo, nullptr, out,
                                              MTOT, DMODEL, DMODEL);
}
